// DocREModel_84284438217062
// MI455X (gfx1250) — matrix operand order repaired and checked by reading, not compiled, not run on hardware
//
#include <hip/hip_runtime.h>
#include <hip/hip_bf16.h>

// ---- problem dimensions (match reference) ----
#define Bn   4
#define Ln   1024
#define Dn   768
#define Hn   12
#define En   42
#define Mn   8
#define Pn   1722
#define EMBn 768
#define BSn  64
#define NLn  97
#define Kgn  12          // EMB / BS groups
#define BPn  (Bn*Pn)     // 6888
#define KBIL (Kgn*BSn*BSn)   // 49152 virtual K of bilinear GEMM
#define NLPAD 112        // 7 tiles of 16

typedef __attribute__((ext_vector_type(16))) __bf16 v16bf;
typedef __attribute__((ext_vector_type(8)))  __bf16 v8bf;
typedef __attribute__((ext_vector_type(8)))  float  v8f;

// ---------------------------------------------------------------------------
// WMMA helper: D = A(16x32 bf16) x B(32x16 bf16) + C (f32)
// ---------------------------------------------------------------------------
__device__ __forceinline__ v8f wmma_bf16(v16bf a, v16bf b, v8f c) {
  return __builtin_amdgcn_wmma_f32_16x16x32_bf16(
      /*neg_a=*/false, a, /*neg_b=*/false, b,
      /*c_mod=*/(short)0, c, /*reuse_a=*/false, /*reuse_b=*/false);
}

// A-fragment (16x32 bf16) from row-major global, wave32 layout:
// lane m = lane&15, half hi = lane>>4; elems 0..7 -> K = hi*8 + i,
// elems 8..15 -> K = 16 + hi*8 + i.  Two 16B vector loads per lane.
__device__ __forceinline__ v16bf load_a16x32(const __bf16* A, int lda,
                                             int row0, int k0, int lane,
                                             int rowMax) {
  int m = row0 + (lane & 15);
  if (m >= rowMax) m = rowMax - 1;          // clamp tail rows (not stored)
  int hi = lane >> 4;
  const __bf16* p = A + (size_t)m * lda + k0 + hi * 8;
  v8bf x = *(const v8bf*)p;
  v8bf y = *(const v8bf*)(p + 16);
  v16bf r;
#pragma unroll
  for (int i = 0; i < 8; ++i) { r[i] = x[i]; r[8 + i] = y[i]; }
  return r;
}

// ---------------------------------------------------------------------------
// Pack a row-major f32 [K,N] matrix into wave32 B-fragment layout (bf16):
// frag index (kt,nt): 32 lanes x 16 bf16; lane column n = nt*16 + (lane&15);
// elems 0..7 hold K = kt*32 + hi*8 + i, elems 8..15 hold K = kt*32 + 16 + hi*8 + (i-8).
// blockIdx.x over KT*NT, blockIdx.y = batch.
// ---------------------------------------------------------------------------
__global__ void k_pack_b(const float* __restrict__ Bsrc, __bf16* __restrict__ Pdst,
                         int K, int N, int NT, int KT) {
  const float* Bb = Bsrc + (size_t)blockIdx.y * K * N;
  __bf16* Pb = Pdst + (size_t)blockIdx.y * KT * NT * 512;
  int kt = blockIdx.x / NT, nt = blockIdx.x % NT;
  int lane = threadIdx.x;
  int n  = nt * 16 + (lane & 15);
  int hi = lane >> 4;
  __bf16* dst = Pb + ((size_t)(kt * NT + nt) * 32 + lane) * 16;
#pragma unroll
  for (int i = 0; i < 16; ++i) {
    int k = kt * 32 + ((i < 8) ? (hi * 8 + i) : (16 + hi * 8 + (i - 8)));
    float v = (k < K && n < N) ? Bb[(size_t)k * N + n] : 0.0f;
    dst[i] = (__bf16)v;
  }
}

// ---------------------------------------------------------------------------
// Entity embedding: masked logsumexp over M mentions -> bf16 [B,E,D]
// ---------------------------------------------------------------------------
__global__ void k_entity_emb(const float* __restrict__ seq,
                             const int* __restrict__ midx,
                             const unsigned char* __restrict__ mmask,
                             __bf16* __restrict__ e_emb) {
  int be = blockIdx.x;
  int b = be / En;
  int idxs[Mn]; int act[Mn]; int cnt = 0;
#pragma unroll
  for (int m = 0; m < Mn; ++m) {
    idxs[m] = midx[be * Mn + m];
    act[m]  = mmask[be * Mn + m] ? 1 : 0;
    cnt += act[m];
  }
  const float* sb = seq + (size_t)b * Ln * Dn;
  for (int d = threadIdx.x; d < Dn; d += blockDim.x) {
    float mx = -3.402823466e38f;
#pragma unroll
    for (int m = 0; m < Mn; ++m)
      if (act[m]) mx = fmaxf(mx, sb[(size_t)idxs[m] * Dn + d]);
    float s = 0.0f;
#pragma unroll
    for (int m = 0; m < Mn; ++m)
      if (act[m]) s += __expf(sb[(size_t)idxs[m] * Dn + d] - mx);
    float r = (cnt > 0) ? (mx + __logf(s)) : 0.0f;
    e_emb[(size_t)be * Dn + d] = (__bf16)r;
  }
}

// ---------------------------------------------------------------------------
// Entity attention: mean over mentions of attention[b,h,idx,:] -> f32 [B,E,H,L]
// ---------------------------------------------------------------------------
__global__ void k_entity_att(const float* __restrict__ att,
                             const int* __restrict__ midx,
                             const unsigned char* __restrict__ mmask,
                             float* __restrict__ e_att) {
  int id = blockIdx.x;
  int h = id % Hn;
  int e = (id / Hn) % En;
  int b = id / (Hn * En);
  int be = b * En + e;
  int idxs[Mn]; int act[Mn]; int cnt = 0;
#pragma unroll
  for (int m = 0; m < Mn; ++m) {
    idxs[m] = midx[be * Mn + m];
    act[m]  = mmask[be * Mn + m] ? 1 : 0;
    cnt += act[m];
  }
  float inv = 1.0f / fmaxf((float)cnt, 1.0f);
  const float* ab = att + ((size_t)b * Hn + h) * Ln * Ln;
  float* ob = e_att + ((size_t)be * Hn + h) * Ln;
  for (int l = threadIdx.x; l < Ln; l += blockDim.x) {
    float s = 0.0f;
#pragma unroll
    for (int m = 0; m < Mn; ++m)
      if (act[m]) s += ab[(size_t)idxs[m] * Ln + l];
    ob[l] = s * inv;
  }
}

// ---------------------------------------------------------------------------
// Pair attention: ht_att = normalize_L( mean_H h_att*t_att ) -> bf16 [B*P, L]
// ---------------------------------------------------------------------------
__global__ void k_pair_att(const float* __restrict__ e_att,
                           const int* __restrict__ hts,
                           __bf16* __restrict__ Aht) {
  __shared__ float v[Ln];
  __shared__ float red[256];
  int bp = blockIdx.x;
  int b = bp / Pn;
  int h0 = hts[bp * 2 + 0];
  int t0 = hts[bp * 2 + 1];
  const float* ha = e_att + (size_t)(b * En + h0) * Hn * Ln;
  const float* ta = e_att + (size_t)(b * En + t0) * Hn * Ln;
  float lsum = 0.0f;
  for (int l = threadIdx.x; l < Ln; l += blockDim.x) {
    float s = 0.0f;
#pragma unroll
    for (int h = 0; h < Hn; ++h) s += ha[h * Ln + l] * ta[h * Ln + l];
    s *= (1.0f / Hn);
    v[l] = s;
    lsum += s;
  }
  red[threadIdx.x] = lsum;
  __syncthreads();
  for (int o = 128; o > 0; o >>= 1) {
    if (threadIdx.x < o) red[threadIdx.x] += red[threadIdx.x + o];
    __syncthreads();
  }
  float inv = 1.0f / (red[0] + 1e-5f);
  for (int l = threadIdx.x; l < Ln; l += blockDim.x)
    Aht[(size_t)bp * Ln + l] = (__bf16)(v[l] * inv);
}

// ---------------------------------------------------------------------------
// Gather hs/ts into columns [0,768) of the concat A matrices (bf16)
// ---------------------------------------------------------------------------
__global__ void k_gather_pairs(const __bf16* __restrict__ e_emb,
                               const int* __restrict__ hts,
                               __bf16* __restrict__ Ah, __bf16* __restrict__ At) {
  int bp = blockIdx.x;
  int b = bp / Pn;
  const __bf16* eh = e_emb + (size_t)(b * En + hts[bp * 2 + 0]) * Dn;
  const __bf16* et = e_emb + (size_t)(b * En + hts[bp * 2 + 1]) * Dn;
  for (int d = threadIdx.x; d < Dn; d += blockDim.x) {
    Ah[(size_t)bp * (2 * Dn) + d] = eh[d];
    At[(size_t)bp * (2 * Dn) + d] = et[d];
  }
}

// ---------------------------------------------------------------------------
// rs GEMM: [P,1024] x [1024,768] per batch, one wave = 16x64 tile.
// 2x-unrolled ping-pong pipeline: buffer set X loads while set Y feeds WMMA,
// no rotation copies.  Epilogue writes bf16 into cols [768,1536) of both
// concat A matrices.
// ---------------------------------------------------------------------------
__global__ void k_gemm_rs(const __bf16* __restrict__ Aht,
                          const __bf16* __restrict__ packSeq,
                          __bf16* __restrict__ Ah, __bf16* __restrict__ At) {
  int b = blockIdx.z;
  int row0 = blockIdx.x * 16;
  int n0   = blockIdx.y * 64;
  int lane = threadIdx.x;
  const __bf16* A = Aht + (size_t)b * Pn * Ln;
  const v16bf* Bp = (const v16bf*)(packSeq + (size_t)b * (Ln / 32) * (Dn / 16) * 512);
  const int NT = Dn / 16;
  const int KT = Ln / 32;                 // 32 (even)
  int ntb = n0 / 16;
  v8f acc[4] = {};
  v16bf aX, aY, bX[4], bY[4];
  aX = load_a16x32(A, Ln, row0, 0, lane, Pn);
#pragma unroll
  for (int t = 0; t < 4; ++t) bX[t] = Bp[(size_t)(ntb + t) * 32 + lane];
  for (int kt = 0; kt < KT; kt += 2) {
    int k1 = kt + 1;                          // always < KT (KT even)
    int k2 = (kt + 2 < KT) ? kt + 2 : kt;     // clamp on final pair
    if (kt + 4 < KT)
      __builtin_prefetch((const void*)&Bp[(size_t)((kt + 4) * NT + ntb) * 32], 0, 1);
    // phase 1: load Y (k1) while consuming X (kt)
    aY = load_a16x32(A, Ln, row0, k1 * 32, lane, Pn);
#pragma unroll
    for (int t = 0; t < 4; ++t) bY[t] = Bp[(size_t)(k1 * NT + ntb + t) * 32 + lane];
#pragma unroll
    for (int t = 0; t < 4; ++t) acc[t] = wmma_bf16(aX, bX[t], acc[t]);
    // phase 2: load X (k2) while consuming Y (k1)
    aX = load_a16x32(A, Ln, row0, k2 * 32, lane, Pn);
#pragma unroll
    for (int t = 0; t < 4; ++t) bX[t] = Bp[(size_t)(k2 * NT + ntb + t) * 32 + lane];
#pragma unroll
    for (int t = 0; t < 4; ++t) acc[t] = wmma_bf16(aY, bY[t], acc[t]);
  }
  int hi = lane >> 4, nn = lane & 15;
#pragma unroll
  for (int r = 0; r < 8; ++r) {
    int m = row0 + 8 * hi + r;
    if (m < Pn) {
      size_t base = ((size_t)b * Pn + m) * (2 * Dn) + Dn + n0 + nn;
#pragma unroll
      for (int t = 0; t < 4; ++t) {
        __bf16 vv = (__bf16)acc[t][r];
        Ah[base + t * 16] = vv;
        At[base + t * 16] = vv;
      }
    }
  }
}

// ---------------------------------------------------------------------------
// Extractor GEMM: [BP,1536] x [1536,768], epilogue tanh(x + bias) -> bf16
// Same 2x-unrolled ping-pong pipeline.
// ---------------------------------------------------------------------------
__global__ void k_gemm_tanh(const __bf16* __restrict__ A,
                            const __bf16* __restrict__ packW,
                            const float* __restrict__ bias,
                            __bf16* __restrict__ Out, int Mrows) {
  int row0 = blockIdx.x * 16;
  int n0   = blockIdx.y * 64;
  int lane = threadIdx.x;
  const int K = 2 * Dn, NT = Dn / 16, KT = K / 32;   // KT = 48 (even)
  const v16bf* Bp = (const v16bf*)packW;
  int ntb = n0 / 16;
  v8f acc[4] = {};
  v16bf aX, aY, bX[4], bY[4];
  aX = load_a16x32(A, K, row0, 0, lane, Mrows);
#pragma unroll
  for (int t = 0; t < 4; ++t) bX[t] = Bp[(size_t)(ntb + t) * 32 + lane];
  for (int kt = 0; kt < KT; kt += 2) {
    int k1 = kt + 1;
    int k2 = (kt + 2 < KT) ? kt + 2 : kt;
    if (kt + 4 < KT)
      __builtin_prefetch((const void*)&Bp[(size_t)((kt + 4) * NT + ntb) * 32], 0, 1);
    aY = load_a16x32(A, K, row0, k1 * 32, lane, Mrows);
#pragma unroll
    for (int t = 0; t < 4; ++t) bY[t] = Bp[(size_t)(k1 * NT + ntb + t) * 32 + lane];
#pragma unroll
    for (int t = 0; t < 4; ++t) acc[t] = wmma_bf16(aX, bX[t], acc[t]);
    aX = load_a16x32(A, K, row0, k2 * 32, lane, Mrows);
#pragma unroll
    for (int t = 0; t < 4; ++t) bX[t] = Bp[(size_t)(k2 * NT + ntb + t) * 32 + lane];
#pragma unroll
    for (int t = 0; t < 4; ++t) acc[t] = wmma_bf16(aY, bY[t], acc[t]);
  }
  int hi = lane >> 4, nn = lane & 15;
#pragma unroll
  for (int r = 0; r < 8; ++r) {
    int m = row0 + 8 * hi + r;
    if (m < Mrows) {
#pragma unroll
      for (int t = 0; t < 4; ++t) {
        int col = n0 + t * 16 + nn;
        Out[(size_t)m * Dn + col] = (__bf16)tanhf(acc[t][r] + bias[col]);
      }
    }
  }
}

// ---------------------------------------------------------------------------
// Grouped bilinear as GEMM [BP, 49152] x [49152, 97(pad 112)].
// A-fragment generated in registers: a[p, g*4096+c*64+d] = hf[p,g*64+c]*tf[p,g*64+d].
// 4 waves/block split the 1536-step K loop; 2x-unrolled ping-pong on both the
// A-generation inputs and the 7 B fragments.  LDS f32 reduction; bias epilogue.
// ---------------------------------------------------------------------------
__global__ void k_bilinear(const __bf16* __restrict__ hf,
                           const __bf16* __restrict__ tf,
                           const __bf16* __restrict__ packWb,
                           const float* __restrict__ bbil,
                           float* __restrict__ out, int Mrows) {
  __shared__ float lds[4 * 16 * NLPAD];
  int row0 = blockIdx.x * 16;
  int lane = threadIdx.x & 31;
  int w    = threadIdx.x >> 5;
  int hi = lane >> 4, nn = lane & 15;
  int pm = row0 + nn;
  if (pm >= Mrows) pm = Mrows - 1;
  const __bf16* hrow = hf + (size_t)pm * EMBn;
  const __bf16* trow = tf + (size_t)pm * EMBn;
  const v16bf* Bp = (const v16bf*)packWb;
  v8f acc[7] = {};
  const int KSTEPS = KBIL / 32;        // 1536
  const int CH = KSTEPS / 4;           // 384 per wave (even)
  const int tEnd = (w + 1) * CH;
  float hvX, hvY; v8bf xX, yX, xY, yY; v16bf bX[7], bY[7];
  {
    int t = w * CH;
    int g = t >> 7, c = (t >> 1) & 63, db = (t & 1) * 32;
    hvX = (float)hrow[g * BSn + c];
    const __bf16* tp = trow + g * BSn + db + hi * 8;
    xX = *(const v8bf*)tp;
    yX = *(const v8bf*)(tp + 16);
#pragma unroll
    for (int nt = 0; nt < 7; ++nt) bX[nt] = Bp[(size_t)(t * 7 + nt) * 32 + lane];
  }
  for (int t = w * CH; t < tEnd; t += 2) {
    int t1 = t + 1;                          // always < tEnd (CH even)
    int t2 = (t + 2 < tEnd) ? t + 2 : t;
    if (t + 8 < tEnd)
      __builtin_prefetch((const void*)&Bp[(size_t)(t + 8) * 7 * 32], 0, 1);
    // phase 1: load Y (t1) while consuming X (t)
    {
      int g = t1 >> 7, c = (t1 >> 1) & 63, db = (t1 & 1) * 32;
      hvY = (float)hrow[g * BSn + c];
      const __bf16* tp = trow + g * BSn + db + hi * 8;
      xY = *(const v8bf*)tp;
      yY = *(const v8bf*)(tp + 16);
#pragma unroll
      for (int nt = 0; nt < 7; ++nt) bY[nt] = Bp[(size_t)(t1 * 7 + nt) * 32 + lane];
    }
    {
      v16bf a;
#pragma unroll
      for (int i = 0; i < 8; ++i) {
        a[i]     = (__bf16)(hvX * (float)xX[i]);
        a[8 + i] = (__bf16)(hvX * (float)yX[i]);
      }
#pragma unroll
      for (int nt = 0; nt < 7; ++nt) acc[nt] = wmma_bf16(a, bX[nt], acc[nt]);
    }
    // phase 2: load X (t2) while consuming Y (t1)
    {
      int g = t2 >> 7, c = (t2 >> 1) & 63, db = (t2 & 1) * 32;
      hvX = (float)hrow[g * BSn + c];
      const __bf16* tp = trow + g * BSn + db + hi * 8;
      xX = *(const v8bf*)tp;
      yX = *(const v8bf*)(tp + 16);
#pragma unroll
      for (int nt = 0; nt < 7; ++nt) bX[nt] = Bp[(size_t)(t2 * 7 + nt) * 32 + lane];
    }
    {
      v16bf a;
#pragma unroll
      for (int i = 0; i < 8; ++i) {
        a[i]     = (__bf16)(hvY * (float)xY[i]);
        a[8 + i] = (__bf16)(hvY * (float)yY[i]);
      }
#pragma unroll
      for (int nt = 0; nt < 7; ++nt) acc[nt] = wmma_bf16(a, bY[nt], acc[nt]);
    }
  }
#pragma unroll
  for (int nt = 0; nt < 7; ++nt)
#pragma unroll
    for (int r = 0; r < 8; ++r)
      lds[(w * 16 + 8 * hi + r) * NLPAD + nt * 16 + nn] = acc[nt][r];
  __syncthreads();
  for (int o = threadIdx.x; o < 16 * NLPAD; o += blockDim.x) {
    int lm = o / NLPAD, col = o % NLPAD;
    float s = lds[lm * NLPAD + col] + lds[(16 + lm) * NLPAD + col] +
              lds[(32 + lm) * NLPAD + col] + lds[(48 + lm) * NLPAD + col];
    int m = row0 + lm;
    if (m < Mrows && col < NLn) out[(size_t)m * NLn + col] = s + bbil[col];
  }
}

// ---------------------------------------------------------------------------
extern "C" void kernel_launch(void* const* d_in, const int* in_sizes, int n_in,
                              void* d_out, int out_size, void* d_ws, size_t ws_size,
                              hipStream_t stream) {
  (void)in_sizes; (void)n_in; (void)out_size; (void)ws_size;
  const float* seq    = (const float*)d_in[0];   // [B,L,D]
  const float* att    = (const float*)d_in[1];   // [B,H,L,L]
  const float* W_head = (const float*)d_in[2];   // [1536,768]
  const float* b_head = (const float*)d_in[3];   // [768]
  const float* W_tail = (const float*)d_in[4];
  const float* b_tail = (const float*)d_in[5];
  const float* W_bil  = (const float*)d_in[6];   // [49152,97]
  const float* b_bil  = (const float*)d_in[7];   // [97]
  const int*   midx   = (const int*)d_in[8];     // [B,E,M]
  const unsigned char* mmask = (const unsigned char*)d_in[9]; // bool [B,E,M]
  const int*   hts    = (const int*)d_in[10];    // [B,P,2]
  float* out = (float*)d_out;                    // [B*P, NL]

  // ---- workspace carve (256B aligned) ----
  char* wp = (char*)d_ws;
  auto carve = [&](size_t bytes) -> char* {
    char* r = wp;
    wp += (bytes + 255) & ~(size_t)255;
    return r;
  };
  __bf16* e_emb   = (__bf16*)carve((size_t)Bn * En * Dn * 2);
  float*  e_att   = (float*) carve((size_t)Bn * En * Hn * Ln * 4);
  __bf16* Aht     = (__bf16*)carve((size_t)BPn * Ln * 2);
  __bf16* packSeq = (__bf16*)carve((size_t)Bn * (Ln / 32) * (Dn / 16) * 512 * 2);
  __bf16* Ahead   = (__bf16*)carve((size_t)BPn * (2 * Dn) * 2);
  __bf16* Atail   = (__bf16*)carve((size_t)BPn * (2 * Dn) * 2);
  __bf16* packWh  = (__bf16*)carve((size_t)(2 * Dn / 32) * (Dn / 16) * 512 * 2);
  __bf16* packWt  = (__bf16*)carve((size_t)(2 * Dn / 32) * (Dn / 16) * 512 * 2);
  __bf16* hfb     = (__bf16*)carve((size_t)BPn * Dn * 2);
  __bf16* tfb     = (__bf16*)carve((size_t)BPn * Dn * 2);
  __bf16* packWb  = (__bf16*)carve((size_t)(KBIL / 32) * 7 * 512 * 2);

  // 1) entity pooling
  k_entity_emb<<<Bn * En, 256, 0, stream>>>(seq, midx, mmask, e_emb);
  k_entity_att<<<Bn * En * Hn, 256, 0, stream>>>(att, midx, mmask, e_att);

  // 2) pack all B matrices into WMMA fragment layout (bf16)
  k_pack_b<<<dim3((Ln / 32) * (Dn / 16), Bn), 32, 0, stream>>>(seq, packSeq, Ln, Dn, Dn / 16, Ln / 32);
  k_pack_b<<<dim3((2 * Dn / 32) * (Dn / 16), 1), 32, 0, stream>>>(W_head, packWh, 2 * Dn, Dn, Dn / 16, 2 * Dn / 32);
  k_pack_b<<<dim3((2 * Dn / 32) * (Dn / 16), 1), 32, 0, stream>>>(W_tail, packWt, 2 * Dn, Dn, Dn / 16, 2 * Dn / 32);
  k_pack_b<<<dim3((KBIL / 32) * 7, 1), 32, 0, stream>>>(W_bil, packWb, KBIL, NLn, 7, KBIL / 32);

  // 3) pair attention + pair gathers
  k_pair_att<<<BPn, 256, 0, stream>>>(e_att, hts, Aht);
  k_gather_pairs<<<BPn, 256, 0, stream>>>(e_emb, hts, Ahead, Atail);

  // 4) rs GEMM (fused concat write into Ahead/Atail cols 768..1535)
  k_gemm_rs<<<dim3((Pn + 15) / 16, Dn / 64, Bn), 32, 0, stream>>>(Aht, packSeq, Ahead, Atail);

  // 5) head/tail extractor GEMMs with tanh
  k_gemm_tanh<<<dim3((BPn + 15) / 16, Dn / 64), 32, 0, stream>>>(Ahead, packWh, b_head, hfb, BPn);
  k_gemm_tanh<<<dim3((BPn + 15) / 16, Dn / 64), 32, 0, stream>>>(Atail, packWt, b_tail, tfb, BPn);

  // 6) grouped bilinear classifier (dominant GEMM, A generated in registers)
  k_bilinear<<<dim3((BPn + 15) / 16), 128, 0, stream>>>(hfb, tfb, packWb, b_bil, out, BPn);
}
